// CCARM1Model_87814901334541
// MI455X (gfx1250) — compile-verified, matrix operand order checked by reading, not run on hardware
//
#include <hip/hip_runtime.h>
#include <hip/hip_bf16.h>
#include <math.h>

typedef _Float16 v16h __attribute__((ext_vector_type(16)));
typedef _Float16 v8h  __attribute__((ext_vector_type(8)));
typedef float    v8f  __attribute__((ext_vector_type(8)));

#define BN   4096
#define TN   60
#define FN   30
#define HN   256
#define HDN  100
#define HDP  128
#define LATN 256
#define DINP 96            // padded encoder x width (74 -> 96)
#define KD1X 288           // padded decoder-1 x width (279 -> 288)

// ---------------- device helpers ----------------

__device__ __forceinline__ v16h load_v16h(const _Float16* p) {
  const v8h* q = (const v8h*)p;
  v8h lo = q[0];
  v8h hi = q[1];
  v16h r;
#pragma unroll
  for (int i = 0; i < 8; ++i) { r[i] = lo[i]; r[8 + i] = hi[i]; }
  return r;
}

__device__ __forceinline__ float sigf(float x) { return 1.0f / (1.0f + __expf(-x)); }

// ---------------- weight packing: f32 [N][K] -> f16 [4*HDp][Kxp+Khp], zero padded ----------------

__global__ void pack_weights(const float* __restrict__ Wx, int KxR, int KxP,
                             const float* __restrict__ Wh, int KhR, int KhP,
                             const float* __restrict__ bx, const float* __restrict__ bh,
                             int HDr, int HDp,
                             _Float16* __restrict__ Wout, float* __restrict__ bout) {
  const int K  = KxP + KhP;
  const int Np = 4 * HDp;
  const int total = Np * K;
  for (int idx = blockIdx.x * blockDim.x + threadIdx.x; idx < total;
       idx += gridDim.x * blockDim.x) {
    int np = idx / K, k = idx % K;
    int g = np / HDp, j = np % HDp;
    float v = 0.0f;
    if (j < HDr) {
      int nr = g * HDr + j;
      if (k < KxP) {
        if (k < KxR) v = Wx[(size_t)nr * KxR + k];
      } else {
        int kh = k - KxP;
        if (Wh && kh < KhR) v = Wh[(size_t)nr * KhR + kh];
      }
    }
    Wout[idx] = (_Float16)v;
    if (k == 0) bout[np] = (j < HDr) ? (bx[g * HDr + j] + (bh ? bh[g * HDr + j] : 0.0f)) : 0.0f;
  }
}

// ---------------- embedding concat: build time-major s[T][B][96] f16 ----------------

__global__ void build_seq(const float* __restrict__ seq, const int* __restrict__ ymd,
                          const int* __restrict__ acq,
                          const float* __restrict__ Ea0, const float* __restrict__ Ea1,
                          const float* __restrict__ Ea2,
                          const float* __restrict__ Es0, const float* __restrict__ Es1,
                          const float* __restrict__ Es2,
                          _Float16* __restrict__ sF) {
  const size_t total = (size_t)BN * TN * DINP;
  for (size_t idx = (size_t)blockIdx.x * blockDim.x + threadIdx.x; idx < total;
       idx += (size_t)gridDim.x * blockDim.x) {
    int k  = (int)(idx % DINP);
    int bt = (int)(idx / DINP);
    int t  = bt % TN;
    int b  = bt / TN;
    float v = 0.0f;
    if      (k < 30) v = seq[((size_t)b * TN + t) * FN + k];
    else if (k < 46) v = Ea0[acq[b * 3 + 0] * 16 + (k - 30)];
    else if (k < 54) v = Ea1[acq[b * 3 + 1] * 8  + (k - 46)];
    else if (k < 58) v = Ea2[acq[b * 3 + 2] * 4  + (k - 54)];
    else if (k < 66) v = Es0[ymd[((size_t)b * TN + t) * 3 + 0] * 8 + (k - 58)];
    else if (k < 70) v = Es1[ymd[((size_t)b * TN + t) * 3 + 1] * 4 + (k - 66)];
    else if (k < 74) v = Es2[ymd[((size_t)b * TN + t) * 3 + 2] * 4 + (k - 70)];
    sF[((size_t)t * BN + b) * DINP + k] = (_Float16)v;
  }
}

__global__ void build_rev(const _Float16* __restrict__ sF, const int* __restrict__ lens,
                          _Float16* __restrict__ sB) {
  const size_t total = (size_t)BN * TN * DINP;
  for (size_t idx = (size_t)blockIdx.x * blockDim.x + threadIdx.x; idx < total;
       idx += (size_t)gridDim.x * blockDim.x) {
    int k  = (int)(idx % DINP);
    size_t tb = idx / DINP;
    int b = (int)(tb % BN);
    int t = (int)(tb / BN);
    int src = lens[b] - 1 - t;
    src = src < 0 ? 0 : (src > TN - 1 ? TN - 1 : src);
    sB[idx] = sF[((size_t)src * BN + b) * DINP + k];
  }
}

// ---------------- fused WMMA LSTM step ----------------
// gates = [x|h] @ [Wih|Whh]^T + b ; in-register cell update with optional length mask.
// Wave: 32 rows x 16 hidden cols, 4 gates -> 8 WMMA accumulators (B tile reused across
// both row tiles: halves weight traffic per WMMA). Block: 8 waves = 64 rows x 64 hidden.
// K-loop split into x-part and h-part so addresses are affine (no per-chunk selects).

__global__ __launch_bounds__(256)
void lstm_step_wmma(const _Float16* __restrict__ Ax, int Kx,
                    const _Float16* __restrict__ Ah, int Kh,
                    const _Float16* __restrict__ W,     // [4*Hpad][Kx+Kh] f16
                    const float* __restrict__ bias,     // [4*Hpad]
                    const _Float16* __restrict__ h_in,  // prev h [B][Hpad] f16
                    float* __restrict__ c,              // [B][Hpad] f32 (in/out)
                    _Float16* __restrict__ h_out,       // [B][Hpad] f16
                    const int* __restrict__ lens, int t,
                    int Hpad) {
  const int K     = Kx + Kh;
  const int lane  = threadIdx.x & 31;
  const int wid   = threadIdx.x >> 5;
  const int row0  = blockIdx.x * 64 + (wid >> 2) * 32;   // two 16-row tiles per wave
  const int hid0  = blockIdx.y * 64 + (wid & 3) * 16;
  const int l15   = lane & 15;
  const int khalf = (lane >> 4) * 16;   // lanes 0-15: K lower half, 16-31: upper half

  v8f acc[4][2];
#pragma unroll
  for (int g = 0; g < 4; ++g)
#pragma unroll
    for (int rt = 0; rt < 2; ++rt)
#pragma unroll
      for (int i = 0; i < 8; ++i) acc[g][rt][i] = 0.0f;

  const _Float16* axp = Ax + (size_t)(row0 + l15) * Kx + khalf;
  const _Float16* ahp = Ah + (size_t)(row0 + l15) * Kh + khalf;
  const size_t axs = (size_t)16 * Kx;   // 16-row stride in Ax
  const size_t ahs = (size_t)16 * Kh;   // 16-row stride in Ah
  const _Float16* wp0 = W + (size_t)(0 * Hpad + hid0 + l15) * K + khalf;
  const _Float16* wp1 = W + (size_t)(1 * Hpad + hid0 + l15) * K + khalf;
  const _Float16* wp2 = W + (size_t)(2 * Hpad + hid0 + l15) * K + khalf;
  const _Float16* wp3 = W + (size_t)(3 * Hpad + hid0 + l15) * K + khalf;

  // x-part of K
  for (int kc = 0; kc < Kx; kc += 32) {
    v16h a0 = load_v16h(axp + kc);
    v16h a1 = load_v16h(axp + axs + kc);
    const _Float16* wg[4] = {wp0 + kc, wp1 + kc, wp2 + kc, wp3 + kc};
#pragma unroll
    for (int g = 0; g < 4; ++g) {
      v16h b = load_v16h(wg[g]);
      acc[g][0] = __builtin_amdgcn_wmma_f32_16x16x32_f16(false, a0, false, b, (short)0,
                                                         acc[g][0], false, false);
      acc[g][1] = __builtin_amdgcn_wmma_f32_16x16x32_f16(false, a1, false, b, (short)0,
                                                         acc[g][1], false, false);
    }
  }
  // h-part of K
  for (int kc = 0; kc < Kh; kc += 32) {
    v16h a0 = load_v16h(ahp + kc);
    v16h a1 = load_v16h(ahp + ahs + kc);
    const _Float16* wg[4] = {wp0 + Kx + kc, wp1 + Kx + kc, wp2 + Kx + kc, wp3 + Kx + kc};
#pragma unroll
    for (int g = 0; g < 4; ++g) {
      v16h b = load_v16h(wg[g]);
      acc[g][0] = __builtin_amdgcn_wmma_f32_16x16x32_f16(false, a0, false, b, (short)0,
                                                         acc[g][0], false, false);
      acc[g][1] = __builtin_amdgcn_wmma_f32_16x16x32_f16(false, a1, false, b, (short)0,
                                                         acc[g][1], false, false);
    }
  }

  const int n = hid0 + l15;
  const float bi = bias[0 * Hpad + n];
  const float bf = bias[1 * Hpad + n];
  const float bg = bias[2 * Hpad + n];
  const float bo = bias[3 * Hpad + n];
  const bool masked = (lens != nullptr);

#pragma unroll
  for (int rt = 0; rt < 2; ++rt) {
#pragma unroll
    for (int r = 0; r < 8; ++r) {
      const int row = row0 + rt * 16 + r + ((lane >> 4) << 3);  // C/D layout: VGPR r -> M=r / r+8
      const size_t idx = (size_t)row * Hpad + n;
      const float gi = acc[0][rt][r] + bi;
      const float gf = acc[1][rt][r] + bf;
      const float gg = acc[2][rt][r] + bg;
      const float go = acc[3][rt][r] + bo;
      const float cold = c[idx];
      float cn = sigf(gf) * cold + sigf(gi) * tanhf(gg);
      float hn = sigf(go) * tanhf(cn);
      if (masked && !(t < lens[row])) { cn = cold; hn = (float)h_in[idx]; }
      c[idx] = cn;
      h_out[idx] = (_Float16)hn;
    }
  }
}

// ---------------- WMMA linear: out[M][N] = A[M][K] @ W[N][K]^T + b ----------------

__global__ __launch_bounds__(256)
void linear_wmma(const _Float16* __restrict__ A, int K,
                 const _Float16* __restrict__ W, const float* __restrict__ bias,
                 float* __restrict__ out, int N) {
  const int lane  = threadIdx.x & 31;
  const int wid   = threadIdx.x >> 5;
  const int row0  = blockIdx.x * 32 + (wid >> 2) * 16;
  const int col0  = blockIdx.y * 64 + (wid & 3) * 16;
  const int l15   = lane & 15;
  const int khalf = (lane >> 4) * 16;
  const int col   = col0 + l15;

  v8f acc;
#pragma unroll
  for (int i = 0; i < 8; ++i) acc[i] = 0.0f;

  const _Float16* ap = A + (size_t)(row0 + l15) * K + khalf;
  const _Float16* wp = W + (size_t)col * K + khalf;
  for (int kc = 0; kc < K; kc += 32) {
    v16h a = load_v16h(ap + kc);
    v16h b = load_v16h(wp + kc);
    acc = __builtin_amdgcn_wmma_f32_16x16x32_f16(false, a, false, b, (short)0, acc,
                                                 false, false);
  }
  const float bs = bias[col];
#pragma unroll
  for (int r = 0; r < 8; ++r) {
    const int row = row0 + r + ((lane >> 4) << 3);
    out[(size_t)row * N + col] = acc[r] + bs;
  }
}

// ---------------- small elementwise / head kernels ----------------

__global__ void add_h16(const _Float16* a, const _Float16* b, _Float16* o, int n) {
  int i = blockIdx.x * blockDim.x + threadIdx.x;
  if (i < n) o[i] = a[i] + b[i];
}

__global__ void make_z(const float* __restrict__ lat, const float* __restrict__ noise,
                       _Float16* __restrict__ A1x) {
  int i = blockIdx.x * blockDim.x + threadIdx.x;
  if (i >= BN * LATN) return;
  int b = i / LATN, k = i % LATN;
  float mu = lat[(size_t)b * 512 + k];
  float lv = lat[(size_t)b * 512 + LATN + k];
  A1x[(size_t)b * KD1X + k] = (_Float16)(mu + sqrtf(__expf(lv)) * noise[i]);
}

__global__ void init_dlq(const float* __restrict__ seq, const int* __restrict__ lens,
                         _Float16* __restrict__ A1x) {
  int i = blockIdx.x * blockDim.x + threadIdx.x;
  if (i >= BN * 9) return;
  int b = i / 9, j = i % 9;
  A1x[(size_t)b * KD1X + 256 + j] =
      (_Float16)seq[((size_t)b * TN + (lens[b] - 1)) * FN + 21 + j];
}

__global__ void set_mp(const float* __restrict__ mp, int t, _Float16* __restrict__ A1x) {
  int i = blockIdx.x * blockDim.x + threadIdx.x;
  if (i >= BN * 14) return;
  int b = i / 14, j = i % 14;
  A1x[(size_t)b * KD1X + 265 + j] = (_Float16)mp[((size_t)b * 12 + t) * 14 + j];
}

__global__ void g1_kernel(const _Float16* __restrict__ h2, const float* __restrict__ Wg1,
                          const float* __restrict__ bg1, float* __restrict__ xg) {
  int i = blockIdx.x * blockDim.x + threadIdx.x;
  if (i >= BN * 25) return;
  int b = i / 25, j = i % 25;
  float a = bg1[j];
  for (int k = 0; k < HDN; ++k) a += (float)h2[(size_t)b * HDP + k] * Wg1[j * HDN + k];
  xg[i] = fmaxf(a, 0.0f);
}

__global__ __launch_bounds__(256)
void bn_stats(const float* __restrict__ xg, float* __restrict__ stats) {
  const int j = blockIdx.x;  // feature 0..24
  float s = 0.0f, s2 = 0.0f;
  for (int b = threadIdx.x; b < BN; b += 256) {
    float v = xg[(size_t)b * 25 + j];
    s += v; s2 += v * v;
  }
  __shared__ float sh[256], sh2[256];
  sh[threadIdx.x] = s; sh2[threadIdx.x] = s2;
  __syncthreads();
  for (int o = 128; o > 0; o >>= 1) {
    if (threadIdx.x < o) { sh[threadIdx.x] += sh[threadIdx.x + o]; sh2[threadIdx.x] += sh2[threadIdx.x + o]; }
    __syncthreads();
  }
  if (threadIdx.x == 0) {
    float m = sh[0] / (float)BN;
    float var = sh2[0] / (float)BN - m * m;
    stats[j]      = m;
    stats[32 + j] = rsqrtf(var + 1e-5f);
  }
}

__global__ void g2_kernel(const float* __restrict__ xg, const float* __restrict__ stats,
                          const float* __restrict__ gamma, const float* __restrict__ beta,
                          const float* __restrict__ Wg2, const float* __restrict__ bg2,
                          int t, float* __restrict__ out, _Float16* __restrict__ A1x) {
  int i = blockIdx.x * blockDim.x + threadIdx.x;
  if (i >= BN * 9) return;
  int b = i / 9, jo = i % 9;
  float a = bg2[jo];
  for (int j = 0; j < 25; ++j) {
    float xn = gamma[j] * (xg[(size_t)b * 25 + j] - stats[j]) * stats[32 + j] + beta[j];
    a += xn * Wg2[jo * 25 + j];
  }
  out[(size_t)b * 108 + jo * 12 + t] = a;            // [B, 9, 12]
  A1x[(size_t)b * KD1X + 256 + jo] = (_Float16)a;    // feed next decoder step
}

// ---------------- host orchestration ----------------

extern "C" void kernel_launch(void* const* d_in, const int* in_sizes, int n_in,
                              void* d_out, int out_size, void* d_ws, size_t ws_size,
                              hipStream_t stream) {
  (void)in_sizes; (void)n_in; (void)out_size; (void)ws_size;
  const float* seq     = (const float*)d_in[0];
  const int*   lens    = (const int*)d_in[1];
  const int*   ymd     = (const int*)d_in[2];
  const int*   acq     = (const int*)d_in[3];
  const float* macro   = (const float*)d_in[4];
  const float* znoise  = (const float*)d_in[5];
  const float* Ea0 = (const float*)d_in[6],  *Ea1 = (const float*)d_in[7],  *Ea2 = (const float*)d_in[8];
  const float* Es0 = (const float*)d_in[9],  *Es1 = (const float*)d_in[10], *Es2 = (const float*)d_in[11];
  const float* WihF = (const float*)d_in[12], *WhhF = (const float*)d_in[13];
  const float* bihF = (const float*)d_in[14], *bhhF = (const float*)d_in[15];
  const float* WihB = (const float*)d_in[16], *WhhB = (const float*)d_in[17];
  const float* bihB = (const float*)d_in[18], *bhhB = (const float*)d_in[19];
  const float* Wlat = (const float*)d_in[20], *blat = (const float*)d_in[21];
  const float* Wih1 = (const float*)d_in[22], *Whh1 = (const float*)d_in[23];
  const float* bih1 = (const float*)d_in[24], *bhh1 = (const float*)d_in[25];
  const float* Wih2 = (const float*)d_in[26], *Whh2 = (const float*)d_in[27];
  const float* bih2 = (const float*)d_in[28], *bhh2 = (const float*)d_in[29];
  const float* Wg1  = (const float*)d_in[30], *bg1  = (const float*)d_in[31];
  const float* gamma = (const float*)d_in[32], *beta = (const float*)d_in[33];
  const float* Wg2  = (const float*)d_in[34], *bg2  = (const float*)d_in[35];
  float* out = (float*)d_out;

  char* p = (char*)d_ws;
  auto carve = [&](size_t bytes) -> void* {
    void* r = (void*)p;
    p += (bytes + 255) & ~(size_t)255;
    return r;
  };
  const int KENC = DINP + HN;          // 352
  const int KDA  = KD1X + HDP;         // 416
  const int KDB  = HDP + HDP;          // 256

  _Float16* sF    = (_Float16*)carve((size_t)TN * BN * DINP * 2);
  _Float16* sB    = (_Float16*)carve((size_t)TN * BN * DINP * 2);
  _Float16* WeF   = (_Float16*)carve((size_t)1024 * KENC * 2);
  float*    beF   = (float*)carve(1024 * 4);
  _Float16* WeB   = (_Float16*)carve((size_t)1024 * KENC * 2);
  float*    beB   = (float*)carve(1024 * 4);
  _Float16* Wl16  = (_Float16*)carve((size_t)512 * 256 * 2);
  float*    bl32  = (float*)carve(512 * 4);
  _Float16* Wd1   = (_Float16*)carve((size_t)512 * KDA * 2);
  float*    bd1   = (float*)carve(512 * 4);
  _Float16* Wd2   = (_Float16*)carve((size_t)512 * KDB * 2);
  float*    bd2   = (float*)carve(512 * 4);
  _Float16* hf[2] = {(_Float16*)carve((size_t)BN * HN * 2), (_Float16*)carve((size_t)BN * HN * 2)};
  _Float16* hb[2] = {(_Float16*)carve((size_t)BN * HN * 2), (_Float16*)carve((size_t)BN * HN * 2)};
  float*    cF    = (float*)carve((size_t)BN * HN * 4);
  float*    cB    = (float*)carve((size_t)BN * HN * 4);
  _Float16* hs    = (_Float16*)carve((size_t)BN * HN * 2);
  float*    lat   = (float*)carve((size_t)BN * 512 * 4);
  _Float16* A1x   = (_Float16*)carve((size_t)BN * KD1X * 2);
  _Float16* h1[2] = {(_Float16*)carve((size_t)BN * HDP * 2), (_Float16*)carve((size_t)BN * HDP * 2)};
  _Float16* h2[2] = {(_Float16*)carve((size_t)BN * HDP * 2), (_Float16*)carve((size_t)BN * HDP * 2)};
  float*    c1    = (float*)carve((size_t)BN * HDP * 4);
  float*    c2    = (float*)carve((size_t)BN * HDP * 4);
  float*    xg    = (float*)carve((size_t)BN * 25 * 4);
  float*    stats = (float*)carve(64 * 4);

  // zero-init state buffers (graph-capture safe)
  hipMemsetAsync(hf[0], 0, (size_t)BN * HN * 2, stream);
  hipMemsetAsync(hb[0], 0, (size_t)BN * HN * 2, stream);
  hipMemsetAsync(cF, 0, (size_t)BN * HN * 4, stream);
  hipMemsetAsync(cB, 0, (size_t)BN * HN * 4, stream);
  hipMemsetAsync(h1[0], 0, (size_t)BN * HDP * 2, stream);
  hipMemsetAsync(h2[0], 0, (size_t)BN * HDP * 2, stream);
  hipMemsetAsync(c1, 0, (size_t)BN * HDP * 4, stream);
  hipMemsetAsync(c2, 0, (size_t)BN * HDP * 4, stream);
  hipMemsetAsync(A1x, 0, (size_t)BN * KD1X * 2, stream);

  // pack weights to padded f16
  pack_weights<<<1408, 256, 0, stream>>>(WihF, 74, DINP, WhhF, HN, HN, bihF, bhhF, HN, HN, WeF, beF);
  pack_weights<<<1408, 256, 0, stream>>>(WihB, 74, DINP, WhhB, HN, HN, bihB, bhhB, HN, HN, WeB, beB);
  pack_weights<<<512, 256, 0, stream>>>(Wlat, 256, 256, nullptr, 0, 0, blat, nullptr, 128, 128, Wl16, bl32);
  pack_weights<<<832, 256, 0, stream>>>(Wih1, 279, KD1X, Whh1, HDN, HDP, bih1, bhh1, HDN, HDP, Wd1, bd1);
  pack_weights<<<512, 256, 0, stream>>>(Wih2, HDN, HDP, Whh2, HDN, HDP, bih2, bhh2, HDN, HDP, Wd2, bd2);

  // embeddings + reversed sequence
  {
    int grid = (int)(((size_t)BN * TN * DINP + 255) / 256);
    build_seq<<<grid, 256, 0, stream>>>(seq, ymd, acq, Ea0, Ea1, Ea2, Es0, Es1, Es2, sF);
    build_rev<<<grid, 256, 0, stream>>>(sF, lens, sB);
  }

  // bidirectional encoder LSTM: 60 fused WMMA steps per direction
  dim3 egrid(BN / 64, HN / 64);   // (64, 4)
  for (int t = 0; t < TN; ++t) {
    const _Float16* xs = sF + (size_t)t * BN * DINP;
    lstm_step_wmma<<<egrid, 256, 0, stream>>>(xs, DINP, hf[t & 1], HN, WeF, beF,
                                              hf[t & 1], cF, hf[(t + 1) & 1], lens, t, HN);
  }
  for (int t = 0; t < TN; ++t) {
    const _Float16* xs = sB + (size_t)t * BN * DINP;
    lstm_step_wmma<<<egrid, 256, 0, stream>>>(xs, DINP, hb[t & 1], HN, WeB, beB,
                                              hb[t & 1], cB, hb[(t + 1) & 1], lens, t, HN);
  }
  _Float16* hFfin = hf[TN & 1];   // TN=60 even -> hf[0]
  _Float16* hBfin = hb[TN & 1];

  // latent: lat = (h_f + h_b) @ W_lat^T + b_lat ; z = mu + sqrt(exp(logv))*noise
  add_h16<<<(BN * HN + 255) / 256, 256, 0, stream>>>(hFfin, hBfin, hs, BN * HN);
  dim3 lgrid(BN / 32, 512 / 64);
  linear_wmma<<<lgrid, 256, 0, stream>>>(hs, 256, Wl16, bl32, lat, 512);
  make_z<<<(BN * LATN + 255) / 256, 256, 0, stream>>>(lat, znoise, A1x);
  init_dlq<<<(BN * 9 + 255) / 256, 256, 0, stream>>>(seq, lens, A1x);

  // decoder: 12 steps of 2-layer WMMA LSTM + batchnorm head
  dim3 dgrid(BN / 64, HDP / 64);  // (64, 2)
  for (int t = 0; t < 12; ++t) {
    set_mp<<<(BN * 14 + 255) / 256, 256, 0, stream>>>(macro, t, A1x);
    lstm_step_wmma<<<dgrid, 256, 0, stream>>>(A1x, KD1X, h1[t & 1], HDP, Wd1, bd1,
                                              h1[t & 1], c1, h1[(t + 1) & 1], nullptr, 0, HDP);
    lstm_step_wmma<<<dgrid, 256, 0, stream>>>(h1[(t + 1) & 1], HDP, h2[t & 1], HDP, Wd2, bd2,
                                              h2[t & 1], c2, h2[(t + 1) & 1], nullptr, 0, HDP);
    g1_kernel<<<(BN * 25 + 255) / 256, 256, 0, stream>>>(h2[(t + 1) & 1], Wg1, bg1, xg);
    bn_stats<<<25, 256, 0, stream>>>(xg, stats);
    g2_kernel<<<(BN * 9 + 255) / 256, 256, 0, stream>>>(xg, stats, gamma, beta, Wg2, bg2,
                                                        t, out, A1x);
  }
}
